// MultiHeadAttention_86294482911783
// MI455X (gfx1250) — matrix operand order repaired and checked by reading, not compiled, not run on hardware
//
#include <hip/hip_runtime.h>
#include <hip/hip_bf16.h>
#include <math.h>

// ---------------- problem constants ----------------
#define N_HEAD 8
#define D_K    256
#define D_MODEL 512
#define SEQ    2048
#define BATCH  2
#define ROWS   (BATCH*SEQ)      // 4096
#define DPROJ  (N_HEAD*D_K)     // 2048
#define LN_EPS 1e-6f

// ---------------- WMMA tile config ----------------
#define TM 128
#define TN 128
#define PITCH 20                // 16 K-pairs (K=32) + pad to 80B: 16B-aligned rows, conflict-free

typedef __attribute__((ext_vector_type(16))) _Float16 v16h;
typedef __attribute__((ext_vector_type(2)))  __fp16   v2fp16;
typedef __attribute__((ext_vector_type(8)))  float    v8f;

union Frag { v16h h; unsigned u[8]; };

__device__ __forceinline__ unsigned pack2f(float a, float b) {
  union { v2fp16 v; unsigned u; } t;
  t.v = __builtin_amdgcn_cvt_pkrtz(a, b);   // v_cvt_pk_rtz_f16_f32
  return t.u;
}

// ---------------- CDNA5 async global->LDS (ASYNCcnt-tracked) ----------------
// LDS byte address = low 32 bits of the generic pointer (ISA 10.2: LDS aperture
// maps addr[31:0] directly; flat pointers to LDS carry the offset in the low word).
__device__ __forceinline__ void async_b128(void* lds_dst, const void* gsrc) {
  unsigned loff = (unsigned)(unsigned long long)lds_dst;
  asm volatile("global_load_async_to_lds_b128 %0, %1, off"
               :: "v"(loff), "v"(gsrc) : "memory");
}
__device__ __forceinline__ void async_wait0() {
  asm volatile("s_wait_asynccnt 0x0" ::: "memory");
}

// Per-wave 32x64 sub-tile of the 128x128 block tile: 2 (M) x 4 (N) wmma tiles.
// LDS holds A as [row][k-pair] and B as [col][k-pair] packed half-pairs.
__device__ __forceinline__ void mma_step(const unsigned (*As)[PITCH],
                                         const unsigned (*Bs)[PITCH],
                                         v8f (&acc)[2][4]) {
  const int lane = threadIdx.x & 31;
  const int wave = threadIdx.x >> 5;
  const int wmb = (wave & 3) * 32;
  const int wnb = (wave >> 2) * 64;
  const int m  = lane & 15;
  const int hi = lane >> 4;
  Frag a[2], b[4];
#pragma unroll
  for (int mt = 0; mt < 2; ++mt) {
    const unsigned* row = As[wmb + mt * 16 + m];
#pragma unroll
    for (int j = 0; j < 8; ++j)      // A 16x32: k = (j>>2)*16 + hi*8 + (j&3)*2
      a[mt].u[j] = row[((j >> 2) << 3) + (hi << 2) + (j & 3)];
  }
#pragma unroll
  for (int nt = 0; nt < 4; ++nt) {
    const unsigned* col = Bs[wnb + nt * 16 + m];
#pragma unroll
    for (int j = 0; j < 8; ++j)      // B 32x16: k = (j>>2)*16 + hi*8 + (j&3)*2
      b[nt].u[j] = col[((j >> 2) << 3) + (hi << 2) + (j & 3)];
  }
#pragma unroll
  for (int mt = 0; mt < 2; ++mt)
#pragma unroll
    for (int nt = 0; nt < 4; ++nt)
      acc[mt][nt] = __builtin_amdgcn_wmma_f32_16x16x32_f16(
          false, a[mt].h, false, b[nt].h, (short)0, acc[mt][nt], false, false);
}

__device__ __forceinline__ void zero_acc(v8f (&acc)[2][4]) {
#pragma unroll
  for (int mt = 0; mt < 2; ++mt)
#pragma unroll
    for (int nt = 0; nt < 4; ++nt)
#pragma unroll
      for (int r = 0; r < 8; ++r) acc[mt][nt][r] = 0.0f;
}

// Stage one 128x32-half operand tile: 2 threads/row, 32B (2x async B128) each.
__device__ __forceinline__ void stage_async(unsigned (*dst)[PITCH],
                                            const _Float16* src, int ld,
                                            int base0, int kt, int t) {
  const int r = t >> 1, h = t & 1;
  const void* g = src + (size_t)(base0 + r) * ld + kt + h * 16;
  async_b128(&dst[r][h * 8],     g);
  async_b128(&dst[r][h * 8 + 4], (const char*)g + 16);
}

// ---------------- generic f16 GEMM: C = A[M,K] @ BT[N,K]^T (+residual) ----------------
__global__ void __launch_bounds__(256) gemm_abt16(
    const _Float16* __restrict__ A, const _Float16* __restrict__ BT,
    const float* __restrict__ residual, float* __restrict__ C,
    int M, int N, int K) {
  __shared__ __align__(16) unsigned As[2][TM][PITCH];
  __shared__ __align__(16) unsigned Bs[2][TN][PITCH];
  const int t = threadIdx.x;
  const int row0 = blockIdx.y * TM;
  const int col0 = blockIdx.x * TN;
  v8f acc[2][4];
  zero_acc(acc);

  stage_async(As[0], A, K, row0, 0, t);
  stage_async(Bs[0], BT, K, col0, 0, t);
  async_wait0();
  __syncthreads();
  for (int kt = 0; kt < K; kt += 32) {
    const int cur = (kt >> 5) & 1;
    if (kt + 32 < K) {
      stage_async(As[cur ^ 1], A, K, row0, kt + 32, t);
      stage_async(Bs[cur ^ 1], BT, K, col0, kt + 32, t);
    }
    mma_step(As[cur], Bs[cur], acc);
    async_wait0();
    __syncthreads();
  }

  const int lane = t & 31, wave = t >> 5;
  const int wmb = (wave & 3) * 32, wnb = (wave >> 2) * 64;
  const int nloc = lane & 15, mhi = (lane >> 4) * 8;
#pragma unroll
  for (int mt = 0; mt < 2; ++mt)
#pragma unroll
    for (int nt = 0; nt < 4; ++nt)
#pragma unroll
      for (int r = 0; r < 8; ++r) {
        int gm = row0 + wmb + mt * 16 + mhi + r;
        int gn = col0 + wnb + nt * 16 + nloc;
        float vv = acc[mt][nt][r];
        if (residual) vv += residual[(size_t)gm * N + gn];
        C[(size_t)gm * N + gn] = vv;
      }
}

// ---------------- scores: S = qh·khᵀ (kh is naturally "BT"), mask fused ----------------
__global__ void __launch_bounds__(256) attn_scores(
    const _Float16* __restrict__ qh, const _Float16* __restrict__ kh,
    const int* __restrict__ mask, float* __restrict__ attn) {
  __shared__ __align__(16) unsigned As[2][TM][PITCH];
  __shared__ __align__(16) unsigned Bs[2][TN][PITCH];
  const int t = threadIdx.x;
  const int bh = blockIdx.z, b = bh >> 3;
  const int row0 = blockIdx.y * TM;   // query
  const int col0 = blockIdx.x * TN;   // key
  const _Float16* Q  = qh + (size_t)bh * SEQ * D_K;
  const _Float16* Kh = kh + (size_t)bh * SEQ * D_K;
  v8f acc[2][4];
  zero_acc(acc);

  stage_async(As[0], Q, D_K, row0, 0, t);
  stage_async(Bs[0], Kh, D_K, col0, 0, t);
  async_wait0();
  __syncthreads();
  for (int kt = 0; kt < D_K; kt += 32) {
    const int cur = (kt >> 5) & 1;
    if (kt + 32 < D_K) {
      stage_async(As[cur ^ 1], Q, D_K, row0, kt + 32, t);
      stage_async(Bs[cur ^ 1], Kh, D_K, col0, kt + 32, t);
    }
    mma_step(As[cur], Bs[cur], acc);
    async_wait0();
    __syncthreads();
  }

  const int lane = t & 31, wave = t >> 5;
  const int wmb = (wave & 3) * 32, wnb = (wave >> 2) * 64;
  const int nloc = lane & 15, mhi = (lane >> 4) * 8;
#pragma unroll
  for (int mt = 0; mt < 2; ++mt)
#pragma unroll
    for (int nt = 0; nt < 4; ++nt)
#pragma unroll
      for (int r = 0; r < 8; ++r) {
        int gm = row0 + wmb + mt * 16 + mhi + r;   // query idx
        int gn = col0 + wnb + nt * 16 + nloc;      // key idx
        float vv = acc[mt][nt][r];
        int mk = mask[((size_t)b * SEQ + gm) * SEQ + gn];
        vv = mk ? vv : -INFINITY;
        attn[((size_t)bh * SEQ + gm) * SEQ + gn] = vv;
      }
}

// ---------------- row softmax over L=2048 (nan-fill fully masked rows) ----------------
__global__ void __launch_bounds__(256) softmax_rows(float* __restrict__ attn) {
  float* p = attn + (size_t)blockIdx.x * SEQ;
  const int t = threadIdx.x;
  float vals[8];
  float lmax = -INFINITY;
#pragma unroll
  for (int j = 0; j < 8; ++j) { vals[j] = p[t + j * 256]; lmax = fmaxf(lmax, vals[j]); }
  __shared__ float red[256];
  red[t] = lmax; __syncthreads();
  for (int s = 128; s > 0; s >>= 1) { if (t < s) red[t] = fmaxf(red[t], red[t + s]); __syncthreads(); }
  float rowmax = red[0];
  __syncthreads();
  if (rowmax == -INFINITY) {     // fully masked row -> zeros (reference nan-fill)
#pragma unroll
    for (int j = 0; j < 8; ++j) p[t + j * 256] = 0.0f;
    return;
  }
  float lsum = 0.0f;
#pragma unroll
  for (int j = 0; j < 8; ++j) { vals[j] = expf(vals[j] - rowmax); lsum += vals[j]; }
  red[t] = lsum; __syncthreads();
  for (int s = 128; s > 0; s >>= 1) { if (t < s) red[t] += red[t + s]; __syncthreads(); }
  float inv = 1.0f / red[0];
  __syncthreads();
#pragma unroll
  for (int j = 0; j < 8; ++j) p[t + j * 256] = vals[j] * inv;
}

// ---------------- ctx = attn(f32, pk-cvt in LDS) @ vhT^T ----------------
__global__ void __launch_bounds__(256) ctx_gemm(
    const float* __restrict__ attn, const _Float16* __restrict__ vhT,
    _Float16* __restrict__ ctx) {
  __shared__ __align__(16) unsigned As[2][TM][PITCH];
  __shared__ __align__(16) unsigned Bs[2][TN][PITCH];
  const int t = threadIdx.x;
  const int bh = blockIdx.z, b = bh >> 3, h = bh & 7;
  const int row0 = blockIdx.y * TM;   // query
  const int col0 = blockIdx.x * TN;   // d_v tile (0 or 128)
  const float*    Ab = attn + (size_t)bh * SEQ * SEQ;
  const _Float16* V  = vhT + (size_t)bh * D_K * SEQ;   // [d][l] ("BT")
  v8f acc[2][4];
  zero_acc(acc);

  const int a_r = t >> 1, a_half = t & 1;
  // prologue stage of buffer 0
  {
    const float* sa = Ab + (size_t)(row0 + a_r) * SEQ + a_half * 16;
#pragma unroll
    for (int i = 0; i < 8; ++i)
      As[0][a_r][a_half * 8 + i] = pack2f(sa[2 * i], sa[2 * i + 1]);
    stage_async(Bs[0], V, SEQ, col0, 0, t);
  }
  async_wait0();
  __syncthreads();
  for (int kt = 0; kt < SEQ; kt += 32) {
    const int cur = (kt >> 5) & 1;
    if (kt + 32 < SEQ) {
      const float* sa = Ab + (size_t)(row0 + a_r) * SEQ + kt + 32 + a_half * 16;
#pragma unroll
      for (int i = 0; i < 8; ++i)
        As[cur ^ 1][a_r][a_half * 8 + i] = pack2f(sa[2 * i], sa[2 * i + 1]);
      stage_async(Bs[cur ^ 1], V, SEQ, col0, kt + 32, t);
      __builtin_prefetch(sa + 32, 0, 1);
    }
    mma_step(As[cur], Bs[cur], acc);
    async_wait0();
    __syncthreads();
  }

  const int lane = t & 31, wave = t >> 5;
  const int wmb = (wave & 3) * 32, wnb = (wave >> 2) * 64;
  const int nloc = lane & 15, mhi = (lane >> 4) * 8;
#pragma unroll
  for (int mt = 0; mt < 2; ++mt)
#pragma unroll
    for (int nt = 0; nt < 4; ++nt)
#pragma unroll
      for (int r = 0; r < 8; ++r) {
        int gm = row0 + wmb + mt * 16 + mhi + r;
        int gn = col0 + wnb + nt * 16 + nloc;
        // ctx layout [b][l][h*256+d] for the output projection GEMM
        ctx[((size_t)b * SEQ + gm) * DPROJ + h * D_K + gn] =
            (_Float16)acc[mt][nt][r];
      }
}

// ---------------- elementwise / transpose helpers ----------------
__global__ void __launch_bounds__(256) cvt_f32_f16(
    const float* __restrict__ src, _Float16* __restrict__ dst, unsigned n) {
  unsigned i = blockIdx.x * 256 + threadIdx.x;
  if (i < n) dst[i] = (_Float16)src[i];
}

// W f32 [K][N] -> WT f16 [N][K]  (one-time; makes every GEMM consume "BT")
__global__ void __launch_bounds__(256) transpose_cvt(
    const float* __restrict__ W, _Float16* __restrict__ WT, int K, int N) {
  __shared__ float tile[32][33];
  const int bx = blockIdx.x * 32;  // N offset
  const int by = blockIdx.y * 32;  // K offset
  const int tx = threadIdx.x & 31, ty = threadIdx.x >> 5;
#pragma unroll
  for (int r = 0; r < 32; r += 8)
    tile[ty + r][tx] = W[(size_t)(by + ty + r) * N + bx + tx];
  __syncthreads();
#pragma unroll
  for (int r = 0; r < 32; r += 8)
    WT[(size_t)(bx + ty + r) * K + by + tx] = (_Float16)tile[tx][ty + r];
}

// proj [b,l,h,d] f32 -> rope -> [b,h,l,d] f16 (scale folds in 1/sqrt(D_K) for q)
__global__ void __launch_bounds__(128) rope_reshape(
    const float* __restrict__ proj, _Float16* __restrict__ dst, float scale) {
  const int bi = blockIdx.x;          // (b*SEQ + l)*8 + h
  const int h = bi & 7;
  const int bl = bi >> 3;             // b*SEQ + l
  const int l = bl & (SEQ - 1);
  const int b = bl >> 11;
  const int i = threadIdx.x;          // pair 0..127
  const float* src = proj + (size_t)bl * DPROJ + h * D_K + 2 * i;
  float x0 = src[0], x1 = src[1];
  float theta = powf(10000.0f, -(float)i * (1.0f / 128.0f));
  float ang = (float)l * theta;
  float s, c;
  sincosf(ang, &s, &c);
  _Float16* d = dst + ((size_t)(b * N_HEAD + h) * SEQ + l) * D_K + 2 * i;
  d[0] = (_Float16)((x0 * c - x1 * s) * scale);
  d[1] = (_Float16)((x1 * c + x0 * s) * scale);
}

// proj [b,l,h,d] f32 -> vhT [b,h,d,l] f16 (transposed so ctx GEMM gets "BT")
__global__ void __launch_bounds__(256) reshape_v_t(
    const float* __restrict__ proj, _Float16* __restrict__ vhT) {
  __shared__ float tile[32][33];
  const int bh = blockIdx.z, b = bh >> 3, h = bh & 7;
  const int l0 = blockIdx.y * 32;
  const int d0 = blockIdx.x * 32;
  const int tx = threadIdx.x & 31, ty = threadIdx.x >> 5;
  const float* src = proj + (size_t)b * SEQ * DPROJ + h * D_K;
#pragma unroll
  for (int r = 0; r < 32; r += 8)
    tile[ty + r][tx] = src[(size_t)(l0 + ty + r) * DPROJ + d0 + tx];
  __syncthreads();
  _Float16* dst = vhT + (size_t)bh * D_K * SEQ;
#pragma unroll
  for (int r = 0; r < 32; r += 8)
    dst[(size_t)(d0 + ty + r) * SEQ + l0 + tx] = (_Float16)tile[tx][ty + r];
}

// ---------------- layernorm over D_MODEL=512 ----------------
__global__ void __launch_bounds__(256) layernorm_rows(
    const float* __restrict__ x, const float* __restrict__ gamma,
    const float* __restrict__ beta, float* __restrict__ out) {
  const float* p = x + (size_t)blockIdx.x * D_MODEL;
  const int t = threadIdx.x;
  float x0 = p[t], x1 = p[t + 256];
  __shared__ float rs[256], rq[256];
  rs[t] = x0 + x1;
  rq[t] = x0 * x0 + x1 * x1;
  __syncthreads();
  for (int s = 128; s > 0; s >>= 1) {
    if (t < s) { rs[t] += rs[t + s]; rq[t] += rq[t + s]; }
    __syncthreads();
  }
  float mean = rs[0] * (1.0f / D_MODEL);
  float var  = rq[0] * (1.0f / D_MODEL) - mean * mean;
  float rstd = rsqrtf(var + LN_EPS);
  float* o = out + (size_t)blockIdx.x * D_MODEL;
  o[t]       = (x0 - mean) * rstd * gamma[t] + beta[t];
  o[t + 256] = (x1 - mean) * rstd * gamma[t + 256] + beta[t + 256];
}

// ---------------- host orchestration ----------------
extern "C" void kernel_launch(void* const* d_in, const int* in_sizes, int n_in,
                              void* d_out, int out_size, void* d_ws, size_t ws_size,
                              hipStream_t stream) {
  (void)in_sizes; (void)n_in; (void)out_size; (void)ws_size;
  const float* q     = (const float*)d_in[0];
  const float* k     = (const float*)d_in[1];
  const float* v     = (const float*)d_in[2];
  const int*   mask  = (const int*)d_in[3];
  const float* Wq    = (const float*)d_in[4];
  const float* Wk    = (const float*)d_in[5];
  const float* Wv    = (const float*)d_in[6];
  const float* Wo    = (const float*)d_in[7];
  const float* gamma = (const float*)d_in[8];
  const float* beta  = (const float*)d_in[9];

  float* out  = (float*)d_out;                 // [4096, 512]
  float* attn = out + (size_t)ROWS * D_MODEL;  // [16, 2048, 2048]

  char* w = (char*)d_ws;
  auto carve = [&](size_t bytes) {
    char* p = w;
    w += (bytes + 255) & ~(size_t)255;
    return p;
  };
  _Float16* q16   = (_Float16*)carve((size_t)ROWS * D_MODEL * 2);
  _Float16* k16   = (_Float16*)carve((size_t)ROWS * D_MODEL * 2);
  _Float16* v16   = (_Float16*)carve((size_t)ROWS * D_MODEL * 2);
  _Float16* WqT   = (_Float16*)carve((size_t)D_MODEL * DPROJ * 2);  // [DPROJ][D_MODEL]
  _Float16* WkT   = (_Float16*)carve((size_t)D_MODEL * DPROJ * 2);
  _Float16* WvT   = (_Float16*)carve((size_t)D_MODEL * DPROJ * 2);
  _Float16* WoT   = (_Float16*)carve((size_t)DPROJ * D_MODEL * 2);  // [D_MODEL][DPROJ]
  float*    proj  = (float*)carve((size_t)ROWS * DPROJ * 4);        // reused q/k/v
  _Float16* qh16  = (_Float16*)carve((size_t)BATCH * N_HEAD * SEQ * D_K * 2);
  _Float16* kh16  = (_Float16*)carve((size_t)BATCH * N_HEAD * SEQ * D_K * 2);
  _Float16* vhT   = (_Float16*)carve((size_t)BATCH * N_HEAD * D_K * SEQ * 2);
  _Float16* ctx16 = (_Float16*)carve((size_t)ROWS * DPROJ * 2);
  float*    preln = (float*)carve((size_t)ROWS * D_MODEL * 4);

  const unsigned nqkv = ROWS * D_MODEL;   // 2,097,152

  cvt_f32_f16<<<(nqkv + 255) / 256, 256, 0, stream>>>(q, q16, nqkv);
  cvt_f32_f16<<<(nqkv + 255) / 256, 256, 0, stream>>>(k, k16, nqkv);
  cvt_f32_f16<<<(nqkv + 255) / 256, 256, 0, stream>>>(v, v16, nqkv);
  dim3 gtw(DPROJ / 32, D_MODEL / 32);
  transpose_cvt<<<gtw, 256, 0, stream>>>(Wq, WqT, D_MODEL, DPROJ);
  transpose_cvt<<<gtw, 256, 0, stream>>>(Wk, WkT, D_MODEL, DPROJ);
  transpose_cvt<<<gtw, 256, 0, stream>>>(Wv, WvT, D_MODEL, DPROJ);
  dim3 gto(D_MODEL / 32, DPROJ / 32);
  transpose_cvt<<<gto, 256, 0, stream>>>(Wo, WoT, DPROJ, D_MODEL);

  dim3 gproj(DPROJ / TN, ROWS / TM);  // (16, 32)
  gemm_abt16<<<gproj, 256, 0, stream>>>(q16, WqT, nullptr, proj, ROWS, DPROJ, D_MODEL);
  rope_reshape<<<ROWS * N_HEAD, 128, 0, stream>>>(proj, qh16, 1.0f / 16.0f);  // / sqrt(D_K)
  gemm_abt16<<<gproj, 256, 0, stream>>>(k16, WkT, nullptr, proj, ROWS, DPROJ, D_MODEL);
  rope_reshape<<<ROWS * N_HEAD, 128, 0, stream>>>(proj, kh16, 1.0f);
  gemm_abt16<<<gproj, 256, 0, stream>>>(v16, WvT, nullptr, proj, ROWS, DPROJ, D_MODEL);
  dim3 gvt(D_K / 32, SEQ / 32, BATCH * N_HEAD);  // (8,64,16)
  reshape_v_t<<<gvt, 256, 0, stream>>>(proj, vhT);

  dim3 gsc(SEQ / TN, SEQ / TM, BATCH * N_HEAD);  // (16,16,16)
  attn_scores<<<gsc, 256, 0, stream>>>(qh16, kh16, mask, attn);
  softmax_rows<<<BATCH * N_HEAD * SEQ, 256, 0, stream>>>(attn);

  dim3 gctx(D_K / TN, SEQ / TM, BATCH * N_HEAD);  // (2,16,16)
  ctx_gemm<<<gctx, 256, 0, stream>>>(attn, vhT, ctx16);

  dim3 gout(D_MODEL / TN, ROWS / TM);  // (4,32)
  gemm_abt16<<<gout, 256, 0, stream>>>(ctx16, WoT, q, preln, ROWS, D_MODEL, DPROJ);

  layernorm_rows<<<ROWS, 256, 0, stream>>>(preln, gamma, beta, out);
}
